// Qwen3_5Attention_75917841924554
// MI455X (gfx1250) — matrix operand order repaired and checked by reading, not compiled, not run on hardware
//
#include <hip/hip_runtime.h>
#include <hip/hip_bf16.h>

// ---------------------------------------------------------------------------
// Qwen3.5-style attention block for gfx1250 (MI455X).
// bf16 WMMA GEMMs with async global->LDS double-buffered staging (ASYNCcnt).
// B=2, S=2048, H=2048, NH=16, NKV=8, HD=128.
// ---------------------------------------------------------------------------

typedef __bf16 bf16_t;
typedef __attribute__((ext_vector_type(16))) __bf16 v16bf;
typedef __attribute__((ext_vector_type(8)))  __bf16 v8bf;
typedef __attribute__((ext_vector_type(8)))  float  v8f;

static constexpr int kB   = 2;
static constexpr int kS   = 2048;
static constexpr int kH   = 2048;
static constexpr int kNH  = 16;
static constexpr int kNKV = 8;
static constexpr int kHD  = 128;
static constexpr int kM   = kB * kS;           // 4096 token rows

// ---------------------------------------------------------------------------
// fp32 -> bf16 conversion
// ---------------------------------------------------------------------------
__global__ __launch_bounds__(256)
void f2bf_kernel(const float* __restrict__ src, bf16_t* __restrict__ dst, long long n)
{
    long long i = (long long)blockIdx.x * 1024 + threadIdx.x * 4;
    if (i + 3 < n) {
        float4 f = *(const float4*)(src + i);
        dst[i + 0] = (bf16_t)f.x;
        dst[i + 1] = (bf16_t)f.y;
        dst[i + 2] = (bf16_t)f.z;
        dst[i + 3] = (bf16_t)f.w;
    } else {
        for (long long j = i; j < n; ++j) dst[j] = (bf16_t)src[j];
    }
}

// ---------------------------------------------------------------------------
// Batched NT GEMM:  C[z][m][n] = alpha * sum_k A[z][m][k] * Bw[z/bdiv][n][k]
// A: bf16 row-major (M x K), Bw: bf16 row-major (N x K), C: fp32 (M x N).
//
// Block tile 128x128x32, 256 threads = 8 wave32s in a 2(M) x 4(N) grid.
// Each wave owns a 64x32 sub-tile: 4 A-fragments x 2 B-fragments = 8 WMMAs
// per K-step. Tiles staged with GLOBAL_LOAD_ASYNC_TO_LDS_B128 (ASYNCcnt),
// double-buffered in LDS (2 x 16 KB).
// All problem sizes here divide the tile sizes exactly (no bounds checks).
// ---------------------------------------------------------------------------
#define BM 128
#define BN 128
#define BK 32

__device__ __forceinline__ unsigned lds_off(const void* p)
{
    // LDS aperture maps addr[31:0] to the LDS byte offset.
    return (unsigned)(uintptr_t)p;
}

__device__ __forceinline__ void async_b128(unsigned lds_addr, const bf16_t* gaddr)
{
    asm volatile("global_load_async_to_lds_b128 %0, %1, off"
                 :: "v"(lds_addr), "v"(gaddr) : "memory");
}
__device__ __forceinline__ void async_b128_off16(unsigned lds_addr, const bf16_t* gaddr)
{
    // IOFFSET is added to BOTH the LDS and global addresses (ISA 15.18.3).
    asm volatile("global_load_async_to_lds_b128 %0, %1, off offset:16"
                 :: "v"(lds_addr), "v"(gaddr) : "memory");
}
__device__ __forceinline__ void wait_async0()
{
    asm volatile("s_wait_asynccnt 0x0" ::: "memory");
}

__global__ __launch_bounds__(256)
void gemm_nt_bf16(const bf16_t* __restrict__ A, const bf16_t* __restrict__ Bw,
                  float* __restrict__ C,
                  int M, int N, int K,
                  long long strideA, long long strideB, long long strideC,
                  int bdiv, float alpha)
{
    // [buf][ A: BM*BK | B: BN*BK ]  halves; 2 x 16 KB
    __shared__ __align__(16) bf16_t sm[2][(BM + BN) * BK];

    const int tid  = threadIdx.x;
    const int lane = tid & 31;
    const int wave = tid >> 5;
    const int wm   = wave >> 2;          // 0..1  (M direction, 64 rows each)
    const int wn   = wave & 3;           // 0..3  (N direction, 32 cols each)

    const int bm = blockIdx.y * BM;
    const int bn = blockIdx.x * BN;
    const int z  = blockIdx.z;

    const bf16_t* Ab = A  + (long long)z * strideA;
    const bf16_t* Bb = Bw + (long long)(z / bdiv) * strideB;
    float*        Cb = C  + (long long)z * strideC;

    v8f acc[4][2] = {};

    // Async staging: each thread moves 32 bytes of A and 32 bytes of B.
    // row = tid/2 (0..127), seg = tid&1 selects 16-half (32B) row segment.
    const int srow = tid >> 1;
    const int sseg = (tid & 1) * 16;

    const unsigned ldsA0 = lds_off(&sm[0][0]);
    const unsigned ldsB0 = ldsA0 + BM * BK * 2;
    const unsigned bufSz = (BM + BN) * BK * 2;

    const bf16_t* gA = Ab + (long long)(bm + srow) * K + sseg;
    const bf16_t* gB = Bb + (long long)(bn + srow) * K + sseg;
    const unsigned laA = (unsigned)((srow * BK + sseg) * 2);
    const unsigned laB = (unsigned)((srow * BK + sseg) * 2);

    auto stage_tile = [&](int buf, int k0) {
        const unsigned base = buf * bufSz;
        const bf16_t* ga = gA + k0;
        const bf16_t* gb = gB + k0;
        async_b128      (ldsA0 + base + laA, ga);
        async_b128_off16(ldsA0 + base + laA, ga);
        async_b128      (ldsB0 + base + laB, gb);
        async_b128_off16(ldsB0 + base + laB, gb);
    };

    const int nk = K / BK;

    // Prologue: stage tile 0.
    stage_tile(0, 0);
    wait_async0();
    __syncthreads();

    const int kg  = lane >> 4;           // K-half select (0/1)
    const int l16 = lane & 15;

    for (int t = 0; t < nk; ++t) {
        const int cur = t & 1;
        if (t + 1 < nk) stage_tile(cur ^ 1, (t + 1) * BK);

        const bf16_t* As = &sm[cur][0];
        const bf16_t* Bs = &sm[cur][BM * BK];

        // A fragments (16x32 MxK): lane m = lane&15; lanes 0-15 hold K 0-7 &
        // 16-23, lanes 16-31 hold K 8-15 & 24-31 (elems 0-7 -> VGPR0-3).
        v16bf afrag[4], bfrag[2];
        #pragma unroll
        for (int i = 0; i < 4; ++i) {
            const bf16_t* ap = As + (wm * 64 + i * 16 + l16) * BK;
            v8bf lo = *(const v8bf*)(ap + kg * 8);
            v8bf hi = *(const v8bf*)(ap + 16 + kg * 8);
            #pragma unroll
            for (int e = 0; e < 8; ++e) { afrag[i][e] = lo[e]; afrag[i][8 + e] = hi[e]; }
        }
        // B fragments (32x16 KxN): lane n = lane&15; lanes 0-15 hold K 0-7 &
        // 16-23, lanes 16-31 hold K 8-15 & 24-31.
        #pragma unroll
        for (int j = 0; j < 2; ++j) {
            const bf16_t* bp = Bs + (wn * 32 + j * 16 + l16) * BK;
            v8bf lo = *(const v8bf*)(bp + kg * 8);
            v8bf hi = *(const v8bf*)(bp + 16 + kg * 8);
            #pragma unroll
            for (int e = 0; e < 8; ++e) { bfrag[j][e] = lo[e]; bfrag[j][8 + e] = hi[e]; }
        }

        #pragma unroll
        for (int i = 0; i < 4; ++i)
            #pragma unroll
            for (int j = 0; j < 2; ++j)
                acc[i][j] = __builtin_amdgcn_wmma_f32_16x16x32_bf16(
                    false, afrag[i], false, bfrag[j],
                    (short)0, acc[i][j], false, false);

        // Drain this iteration's async stores into LDS, then rendezvous.
        wait_async0();
        __syncthreads();
    }

    // Epilogue: C layout lane n = lane&15, m = vgpr + 8*(lane>>4).
    const int mh = (lane >> 4) * 8;
    #pragma unroll
    for (int i = 0; i < 4; ++i)
        #pragma unroll
        for (int j = 0; j < 2; ++j)
            #pragma unroll
            for (int v = 0; v < 8; ++v) {
                const int m = bm + wm * 64 + i * 16 + mh + v;
                const int n = bn + wn * 32 + j * 16 + l16;
                Cb[(long long)m * N + n] = acc[i][j][v] * alpha;
            }
}

// ---------------------------------------------------------------------------
// Fused per-head RMSNorm + RoPE, fp32 in -> bf16 out in (b, head, s, d) layout.
// One block (128 threads) per (b, s, head).
// ---------------------------------------------------------------------------
__global__ __launch_bounds__(128)
void norm_rope_kernel(const float* __restrict__ raw, const float* __restrict__ w,
                      const float* __restrict__ cosp, const float* __restrict__ sinp,
                      bf16_t* __restrict__ dst, int nheads, int rowstride, int headstep)
{
    const int h  = blockIdx.x % nheads;
    const int bs = blockIdx.x / nheads;
    const int b  = bs / kS;
    const int s  = bs % kS;
    const int d  = threadIdx.x;

    __shared__ float red[128];
    __shared__ float xn[128];

    const float x = raw[(size_t)bs * rowstride + h * headstep + d];
    red[d] = x * x;
    __syncthreads();
    #pragma unroll
    for (int off = 64; off > 0; off >>= 1) {
        if (d < off) red[d] += red[d + off];
        __syncthreads();
    }
    const float inv = rsqrtf(red[0] * (1.0f / (float)kHD) + 1e-6f);
    const float v   = x * inv * w[d];
    xn[d] = v;
    __syncthreads();
    const float rot = (d < kHD / 2) ? -xn[d + kHD / 2] : xn[d - kHD / 2];

    const size_t cidx = ((size_t)b * kS + s) * kHD + d;
    const float  o    = v * cosp[cidx] + rot * sinp[cidx];
    dst[(((size_t)b * nheads + h) * kS + s) * kHD + d] = (bf16_t)o;
}

// ---------------------------------------------------------------------------
// V: (b, s, kv, d) fp32 -> (b, kv, d, s) bf16 so PV becomes an NT GEMM.
// ---------------------------------------------------------------------------
__global__ __launch_bounds__(256)
void v_transpose_kernel(const float* __restrict__ vr, bf16_t* __restrict__ vt)
{
    const size_t i = (size_t)blockIdx.x * 256 + threadIdx.x;   // B*S*NKV*HD
    const int    d  = (int)(i % kHD);
    const int    kv = (int)((i / kHD) % kNKV);
    const size_t bs = i / (kHD * kNKV);
    const int    b  = (int)(bs / kS);
    const int    s  = (int)(bs % kS);
    vt[(((size_t)b * kNKV + kv) * kHD + d) * kS + s] = (bf16_t)vr[i];
}

// ---------------------------------------------------------------------------
// Row softmax with the reference's masking: masked = rowmin(raw) - 20.
// In-place on the fp32 attn region of d_out; also emits bf16 copy for PV GEMM.
// One block (256 threads) per row of S=2048.
// ---------------------------------------------------------------------------
__device__ __forceinline__ float blk_red(float v, float* red, int tid, int op)
{
    __syncthreads();
    red[tid] = v;
    __syncthreads();
    #pragma unroll
    for (int off = 128; off > 0; off >>= 1) {
        if (tid < off) {
            const float a = red[tid], b = red[tid + off];
            red[tid] = (op == 0) ? fminf(a, b) : (op == 1) ? fmaxf(a, b) : (a + b);
        }
        __syncthreads();
    }
    return red[0];
}

__global__ __launch_bounds__(256)
void softmax_mask_kernel(float* __restrict__ attn, bf16_t* __restrict__ attn_bf)
{
    __shared__ float red[256];
    const long long row = blockIdx.x;          // B*NH*S rows
    const int q   = (int)(row % kS);
    const int tid = threadIdx.x;
    float* p = attn + row * (long long)kS;

    float v[8];
    #pragma unroll
    for (int j = 0; j < 8; ++j) v[j] = p[tid + j * 256];

    float mn = v[0];
    #pragma unroll
    for (int j = 1; j < 8; ++j) mn = fminf(mn, v[j]);
    const float rowmin = blk_red(mn, red, tid, 0);
    const float masked = rowmin - 20.0f;

    float mx = -3.402823466e38f;
    #pragma unroll
    for (int j = 0; j < 8; ++j) {
        const int k = tid + j * 256;
        if (k > q) v[j] = masked;
        mx = fmaxf(mx, v[j]);
    }
    const float rowmax = blk_red(mx, red, tid, 1);

    float s = 0.0f;
    #pragma unroll
    for (int j = 0; j < 8; ++j) { v[j] = __expf(v[j] - rowmax); s += v[j]; }
    const float inv = 1.0f / blk_red(s, red, tid, 2);

    #pragma unroll
    for (int j = 0; j < 8; ++j) {
        const int   k  = tid + j * 256;
        const float pj = v[j] * inv;
        p[k] = pj;
        attn_bf[row * (long long)kS + k] = (bf16_t)pj;
    }
}

// ---------------------------------------------------------------------------
// Gating: y[b,s,h*HD+d] = ctx[b,h,s,d] * sigmoid(gate[b,s,h,d]) -> bf16.
// gate lives in the second half of each 256-wide head slot of q_raw.
// ---------------------------------------------------------------------------
__global__ __launch_bounds__(256)
void gate_mul_kernel(const float* __restrict__ ctx, const float* __restrict__ q_raw,
                     bf16_t* __restrict__ y)
{
    const size_t i  = (size_t)blockIdx.x * 256 + threadIdx.x;   // B*S*NH*HD
    const int    d  = (int)(i % kHD);
    const int    h  = (int)((i / kHD) % kNH);
    const size_t bs = i / (kHD * kNH);
    const int    b  = (int)(bs / kS);
    const int    s  = (int)(bs % kS);

    const float c = ctx[(((size_t)b * kNH + h) * kS + s) * kHD + d];
    const float g = q_raw[bs * (size_t)(kNH * 2 * kHD) + h * (2 * kHD) + kHD + d];
    y[bs * (size_t)(kNH * kHD) + h * kHD + d] = (bf16_t)(c * (1.0f / (1.0f + __expf(-g))));
}

// ---------------------------------------------------------------------------
// Host-side orchestration (graph-capture safe: only kernel launches on stream).
// ---------------------------------------------------------------------------
extern "C" void kernel_launch(void* const* d_in, const int* in_sizes, int n_in,
                              void* d_out, int out_size, void* d_ws, size_t ws_size,
                              hipStream_t stream)
{
    (void)in_sizes; (void)n_in; (void)out_size; (void)ws_size;

    const float* hidden = (const float*)d_in[0];
    const float* cosp   = (const float*)d_in[1];
    const float* sinp   = (const float*)d_in[2];
    /* d_in[3] = attention_mask: deterministic causal, reproduced in-kernel */
    const float* Wq = (const float*)d_in[4];
    const float* Wk = (const float*)d_in[5];
    const float* Wv = (const float*)d_in[6];
    const float* Wo = (const float*)d_in[7];
    const float* qw = (const float*)d_in[8];
    const float* kw = (const float*)d_in[9];

    float* out_proj = (float*)d_out;                                  // B*S*H
    float* out_attn = (float*)d_out + (size_t)kB * kS * kH;           // B*NH*S*S

    // Workspace carve-out (256B aligned slices).
    char* wp = (char*)d_ws;
    auto carve = [&](size_t bytes) -> char* {
        char* p = wp; wp += (bytes + 255) & ~(size_t)255; return p;
    };
    bf16_t* hid_bf  = (bf16_t*)carve((size_t)kM * kH * 2);
    bf16_t* wq_bf   = (bf16_t*)carve((size_t)kNH * 2 * kHD * kH * 2);
    bf16_t* wk_bf   = (bf16_t*)carve((size_t)kNKV * kHD * kH * 2);
    bf16_t* wv_bf   = (bf16_t*)carve((size_t)kNKV * kHD * kH * 2);
    bf16_t* wo_bf   = (bf16_t*)carve((size_t)kH * kNH * kHD * 2);
    float*  q_raw   = (float*) carve((size_t)kM * kNH * 2 * kHD * 4);
    float*  k_raw   = (float*) carve((size_t)kM * kNKV * kHD * 4);
    float*  v_raw   = (float*) carve((size_t)kM * kNKV * kHD * 4);
    bf16_t* q_bf    = (bf16_t*)carve((size_t)kM * kNH * kHD * 2);     // (b,h,s,d)
    bf16_t* k_bf    = (bf16_t*)carve((size_t)kM * kNKV * kHD * 2);    // (b,kv,s,d)
    bf16_t* v_t     = (bf16_t*)carve((size_t)kM * kNKV * kHD * 2);    // (b,kv,d,s)
    bf16_t* attn_bf = (bf16_t*)carve((size_t)kB * kNH * kS * kS * 2);
    float*  ctx     = (float*) carve((size_t)kB * kNH * kS * kHD * 4);// (b,h,s,d)
    bf16_t* y_bf    = (bf16_t*)carve((size_t)kM * kNH * kHD * 2);

    const dim3 blk256(256);
    auto cvt = [&](const float* s, bf16_t* d, long long n) {
        f2bf_kernel<<<dim3((unsigned)((n + 1023) / 1024)), blk256, 0, stream>>>(s, d, n);
    };

    // 1) bf16 copies of activations and weights.
    cvt(hidden, hid_bf, (long long)kM * kH);
    cvt(Wq, wq_bf, (long long)kNH * 2 * kHD * kH);
    cvt(Wk, wk_bf, (long long)kNKV * kHD * kH);
    cvt(Wv, wv_bf, (long long)kNKV * kHD * kH);
    cvt(Wo, wo_bf, (long long)kH * kNH * kHD);

    // 2) QKV projections (NT GEMMs, fp32 accumulate).
    gemm_nt_bf16<<<dim3(kNH * 2 * kHD / BN, kM / BM, 1), blk256, 0, stream>>>(
        hid_bf, wq_bf, q_raw, kM, kNH * 2 * kHD, kH, 0, 0, 0, 1, 1.0f);
    gemm_nt_bf16<<<dim3(kNKV * kHD / BN, kM / BM, 1), blk256, 0, stream>>>(
        hid_bf, wk_bf, k_raw, kM, kNKV * kHD, kH, 0, 0, 0, 1, 1.0f);
    gemm_nt_bf16<<<dim3(kNKV * kHD / BN, kM / BM, 1), blk256, 0, stream>>>(
        hid_bf, wv_bf, v_raw, kM, kNKV * kHD, kH, 0, 0, 0, 1, 1.0f);

    // 3) RMSNorm + RoPE on Q and K; transpose V.
    norm_rope_kernel<<<dim3(kM * kNH), dim3(128), 0, stream>>>(
        q_raw, qw, cosp, sinp, q_bf, kNH, kNH * 2 * kHD, 2 * kHD);
    norm_rope_kernel<<<dim3(kM * kNKV), dim3(128), 0, stream>>>(
        k_raw, kw, cosp, sinp, k_bf, kNKV, kNKV * kHD, kHD);
    v_transpose_kernel<<<dim3(kM * kNKV * kHD / 256), blk256, 0, stream>>>(v_raw, v_t);

    // 4) Scores: attn[b,h] = scale * Q[b,h] K[b,h/2]^T  (raw, into d_out).
    const float scale = 0.0883883476483184f;   // HD^-0.5
    gemm_nt_bf16<<<dim3(kS / BN, kS / BM, kB * kNH), blk256, 0, stream>>>(
        q_bf, k_bf, out_attn, kS, kS, kHD,
        (long long)kS * kHD, (long long)kS * kHD, (long long)kS * kS, 2, scale);

    // 5) Masking (rowmin-20) + softmax, in place; bf16 copy for PV.
    softmax_mask_kernel<<<dim3(kB * kNH * kS), blk256, 0, stream>>>(out_attn, attn_bf);

    // 6) Context: ctx[b,h] = P[b,h] V[b,h/2]   (NT against V^T).
    gemm_nt_bf16<<<dim3(kHD / BN, kS / BM, kB * kNH), blk256, 0, stream>>>(
        attn_bf, v_t, ctx, kS, kHD, kS,
        (long long)kS * kS, (long long)kHD * kS, (long long)kS * kHD, 2, 1.0f);

    // 7) Sigmoid gating -> bf16 activations for output projection.
    gate_mul_kernel<<<dim3(kM * kNH * kHD / 256), blk256, 0, stream>>>(ctx, q_raw, y_bf);

    // 8) Output projection into d_out.
    gemm_nt_bf16<<<dim3(kH / BN, kM / BM, 1), blk256, 0, stream>>>(
        y_bf, wo_bf, out_proj, kM, kH, kNH * kHD, 0, 0, 0, 1, 1.0f);
}
